// MAB_16449724744103
// MI455X (gfx1250) — matrix operand order repaired and checked by reading, not compiled, not run on hardware
//
#include <hip/hip_runtime.h>
#include <hip/hip_bf16.h>
#include <math.h>

// ---------------------------------------------------------------------------
// Types for CDNA5 WMMA (wave32): A/B = 16 x bf16 per lane, C/D = 8 x f32.
// ---------------------------------------------------------------------------
typedef __attribute__((ext_vector_type(16))) __bf16 v16bf;
typedef __attribute__((ext_vector_type(8)))  __bf16 v8bf;
typedef __attribute__((ext_vector_type(8)))  float  v8f;

union FragBF {
    v16bf  v;
    v8bf   h[2];
    __bf16 e[16];
};

__device__ __forceinline__ __bf16 tobf(float f) { return (__bf16)f; }   // native v_cvt

__device__ __forceinline__ v8f wmma_bf16(v16bf a, v16bf b, v8f c) {
    // D = A(16x32 bf16) * B(32x16 bf16) + C(16x16 f32)
    return __builtin_amdgcn_wmma_f32_16x16x32_bf16(false, a, false, b,
                                                   (short)0, c, false, false);
}

__device__ __forceinline__ float rmax16(float x) {
    #pragma unroll
    for (int o = 1; o < 16; o <<= 1) x = fmaxf(x, __shfl_xor(x, o, 32));
    return x;
}
__device__ __forceinline__ float rsum32(float x) {
    #pragma unroll
    for (int o = 1; o < 32; o <<= 1) x += __shfl_xor(x, o, 32);
    return x;
}

// ---------------------------------------------------------------------------
// Weight transpose + bf16 convert:  W[K][N] (f32) -> Wt[N][K] (bf16)
// ---------------------------------------------------------------------------
__global__ void transpose_bf16(const float* __restrict__ W,
                               unsigned short* __restrict__ Wt,
                               int K, int N) {
    int i = blockIdx.x * 256 + threadIdx.x;
    if (i >= K * N) return;
    int n = i / K;
    int k = i - n * K;
    __bf16* o = (__bf16*)Wt;
    o[i] = tobf(W[(size_t)k * N + n]);
}

// ---------------------------------------------------------------------------
// WMMA GEMM: wave computes 16(M) x 64(N) tile of C = A[M,K] @ Bt[N,K]^T + bias
//   A_IS_F32 selects the A operand path at compile time (branch-free loop).
//   Bt: bf16, [N][K] (weights pre-transposed -> K contiguous per column)
// Epilogue: optional relu, optional residual add (f32 [M,N]),
//           outputs: f32 row-major, bf16 row-major, bf16 transposed-per-batch.
// ---------------------------------------------------------------------------
template <bool A_IS_F32>
__global__ __launch_bounds__(256) void gemm16x64(
    const float* __restrict__ Af32, const unsigned short* __restrict__ Abf,
    const unsigned short* __restrict__ Bt, const float* __restrict__ bias,
    int M, int N, int K,
    float* __restrict__ outF, unsigned short* __restrict__ outB,
    unsigned short* __restrict__ outT, int t_seq,
    const float* __restrict__ residual, int do_relu)
{
    int lane = threadIdx.x & 31;
    int wv   = threadIdx.x >> 5;
    long long wid = (long long)blockIdx.x * 8 + wv;
    int tiles_n = N >> 6;
    long long total = (long long)(M >> 4) * tiles_n;
    if (wid >= total) return;                 // wave-uniform: EXEC stays all-1s
    int       nt = (int)(wid % tiles_n);
    long long mt = wid / tiles_n;
    int m0 = (int)(mt << 4);
    int n0 = nt << 6;
    int half = lane >> 4;
    int lr   = lane & 15;
    size_t arow = (size_t)(m0 + lr) * K;
    const __bf16* Ab = (const __bf16*)Abf;
    const __bf16* Bb = (const __bf16*)Bt;

    v8f zero = {0.f,0.f,0.f,0.f,0.f,0.f,0.f,0.f};
    v8f acc[4] = {zero, zero, zero, zero};

    for (int kk = 0; kk < K; kk += 32) {
        FragBF a;
        if (A_IS_F32) {
            v8f f0 = *(const v8f*)(Af32 + arow + kk + half * 16);
            v8f f1 = *(const v8f*)(Af32 + arow + kk + half * 16 + 8);
            #pragma unroll
            for (int i = 0; i < 8; i++) { a.e[i] = tobf(f0[i]); a.e[8 + i] = tobf(f1[i]); }
        } else {
            a.h[0] = *(const v8bf*)(Ab + arow + kk + half * 16);
            a.h[1] = *(const v8bf*)(Ab + arow + kk + half * 16 + 8);
        }
        #pragma unroll
        for (int t = 0; t < 4; t++) {
            int ncol = n0 + t * 16 + lr;
            v16bf b = *(const v16bf*)(Bb + (size_t)ncol * K + kk + half * 16);
            acc[t] = wmma_bf16(a.v, b, acc[t]);
        }
    }

    __bf16* oB = (__bf16*)outB;
    __bf16* oT = (__bf16*)outT;
    #pragma unroll
    for (int t = 0; t < 4; t++) {
        int col = n0 + t * 16 + lr;
        float bv = bias ? bias[col] : 0.f;
        #pragma unroll
        for (int r = 0; r < 8; r++) {
            size_t rrow = (size_t)m0 + r + half * 8;
            float v = acc[t][r] + bv;
            if (do_relu) v = fmaxf(v, 0.f);
            if (residual) v += residual[rrow * N + col];
            if (outF) outF[rrow * N + col] = v;
            if (oB) oB[rrow * N + col] = tobf(v);
            if (oT) {
                size_t bb  = rrow / t_seq;
                size_t key = rrow - bb * t_seq;
                oT[(bb * N + col) * (size_t)t_seq + key] = tobf(v);
            }
        }
    }
}

// ---------------------------------------------------------------------------
// Flash attention. One wave owns (b, h, 16 q-rows) and streams all 2048 keys
// in 64-key super-chunks:
//   4 WMMAs for S = Qh*Kh^T, one 16-lane max butterfly per row per 64 keys,
//   row sums via WMMA against an all-ones B fragment (no shuffles),
//   P staged through LDS (C-layout -> A-layout), 4 WMMAs for O += P*Vh.
//   Qp:  bf16 [B*NQ][256] row-major
//   Kp:  bf16 [B*NK][256] row-major   (acts as B operand of Q*K^T directly)
//   Vt:  bf16 [B][256][NK]            (transposed -> key-contiguous B operand)
//   Oatt: f32 [B*NQ][256]
// ---------------------------------------------------------------------------
#define ATT_LDSTRIDE 72   // bf16 per P row: 144B -> bank-spread, 16B aligned
__global__ __launch_bounds__(256) void attn_kernel(
    const unsigned short* __restrict__ Qp_, const unsigned short* __restrict__ Kp_,
    const unsigned short* __restrict__ Vt_, const int* __restrict__ mask,
    float* __restrict__ Oatt)
{
    __shared__ __align__(32) __bf16 Plds[8][16 * ATT_LDSTRIDE];
    const __bf16* Qp = (const __bf16*)Qp_;
    const __bf16* Kp = (const __bf16*)Kp_;
    const __bf16* Vt = (const __bf16*)Vt_;

    int lane = threadIdx.x & 31;
    int wv   = threadIdx.x >> 5;
    int wid  = blockIdx.x * 8 + wv;          // 0..4095
    int b  = wid >> 10;                      // / (H * NQ/16)
    int h  = (wid >> 7) & 7;
    int mq = (wid & 127) << 4;
    int half = lane >> 4;
    int lr   = lane & 15;
    __bf16* pl = &Plds[wv][0];

    // Q A-fragment (16x32, dh = head dim 32), fixed for whole key loop
    FragBF aQ;
    {
        size_t qoff = ((size_t)(b * 2048 + mq + lr)) * 256 + h * 32;
        aQ.h[0] = *(const v8bf*)(Qp + qoff + half * 16);
        aQ.h[1] = *(const v8bf*)(Qp + qoff + half * 16 + 8);
    }

    // all-ones B fragment: row-sum reducer on the matrix pipe
    FragBF ones;
    #pragma unroll
    for (int i = 0; i < 16; i++) ones.e[i] = (__bf16)1.0f;

    v8f zero = {0.f,0.f,0.f,0.f,0.f,0.f,0.f,0.f};
    v8f acc0 = zero, acc1 = zero;            // O tile: 16 x 32 (dh 0-15, 16-31)
    v8f lfrag = zero;                        // running softmax denominators
    float mrun[8];
    #pragma unroll
    for (int r = 0; r < 8; r++) mrun[r] = -INFINITY;

    const float scale = 0.0625f;             // 1/sqrt(256) (reference uses dim_V)
    const int*    maskb = mask + b * 2048;
    const __bf16* kbB = Kp + (size_t)b * 2048 * 256 + h * 32;
    const __bf16* vbB = Vt + ((size_t)b * 256 + h * 32) * 2048;

    for (int kc = 0; kc < 2048; kc += 64) {
        // prefetch K/V super-chunk two ahead (128 keys) (wraps to stay in-bounds)
        int kn = (kc + 128) & 2047;
        __builtin_prefetch(kbB + (size_t)(kn + lr) * 256 + half * 16, 0, 1);
        __builtin_prefetch(vbB + (size_t)lr * 2048 + kn + half * 16, 0, 1);

        // ---- scores: four 16x16 key tiles ----
        v8f s[4];
        int mv[4];
        #pragma unroll
        for (int t = 0; t < 4; t++) {
            v16bf bk = *(const v16bf*)(kbB + (size_t)(kc + 16 * t + lr) * 256 + half * 16);
            s[t] = wmma_bf16(aQ.v, bk, zero);
            mv[t] = maskb[kc + 16 * t + lr];
        }

        // ---- online softmax: one max butterfly per row per 64 keys ----
        #pragma unroll
        for (int r = 0; r < 8; r++) {
            float x[4];
            #pragma unroll
            for (int t = 0; t < 4; t++) x[t] = mv[t] ? s[t][r] * scale : -1e9f;
            float xm = fmaxf(fmaxf(x[0], x[1]), fmaxf(x[2], x[3]));
            float mnew = fmaxf(mrun[r], rmax16(xm));
            float corr = __expf(mrun[r] - mnew);
            mrun[r] = mnew;
            int prow = (r + 8 * half) * ATT_LDSTRIDE;
            #pragma unroll
            for (int t = 0; t < 4; t++)
                pl[prow + 16 * t + lr] = tobf(__expf(x[t] - mnew));
            acc0[r]  *= corr;
            acc1[r]  *= corr;
            lfrag[r] *= corr;
        }

        // ---- P A-fragments (16x32 each) from LDS ----
        FragBF aP0, aP1;
        aP0.h[0] = *(const v8bf*)(pl + lr * ATT_LDSTRIDE + half * 16);
        aP0.h[1] = *(const v8bf*)(pl + lr * ATT_LDSTRIDE + half * 16 + 8);
        aP1.h[0] = *(const v8bf*)(pl + lr * ATT_LDSTRIDE + 32 + half * 16);
        aP1.h[1] = *(const v8bf*)(pl + lr * ATT_LDSTRIDE + 40 + half * 16);

        // denominators: l += P @ ones   (row sum in every column slot)
        lfrag = wmma_bf16(aP0.v, ones.v, lfrag);
        lfrag = wmma_bf16(aP1.v, ones.v, lfrag);

        // ---- O += P * V ----
        v16bf bv00 = *(const v16bf*)(vbB + (size_t)lr * 2048 + kc + half * 16);
        v16bf bv01 = *(const v16bf*)(vbB + (size_t)lr * 2048 + kc + 32 + half * 16);
        v16bf bv10 = *(const v16bf*)(vbB + (size_t)(16 + lr) * 2048 + kc + half * 16);
        v16bf bv11 = *(const v16bf*)(vbB + (size_t)(16 + lr) * 2048 + kc + 32 + half * 16);
        acc0 = wmma_bf16(aP0.v, bv00, acc0);
        acc0 = wmma_bf16(aP1.v, bv01, acc0);
        acc1 = wmma_bf16(aP0.v, bv10, acc1);
        acc1 = wmma_bf16(aP1.v, bv11, acc1);
    }

    #pragma unroll
    for (int r = 0; r < 8; r++) {
        float inv = 1.0f / lfrag[r];
        size_t rrow = (size_t)b * 2048 + mq + r + 8 * half;
        Oatt[rrow * 256 + h * 32 + lr]      = acc0[r] * inv;
        Oatt[rrow * 256 + h * 32 + 16 + lr] = acc1[r] * inv;
    }
}

// ---------------------------------------------------------------------------
// LayerNorm over 256 cols, one wave per row. Optional residual add first.
// Writes f32 and optionally a bf16 copy (for the next WMMA GEMM's A operand).
// ---------------------------------------------------------------------------
__global__ __launch_bounds__(256) void ln_kernel(
    const float* __restrict__ X, const float* __restrict__ res,
    const float* __restrict__ g, const float* __restrict__ beta,
    float* __restrict__ outF, unsigned short* __restrict__ outB, int nrows)
{
    int lane = threadIdx.x & 31;
    int wv   = threadIdx.x >> 5;
    long long row = (long long)blockIdx.x * 8 + wv;
    if (row >= nrows) return;
    size_t base = (size_t)row * 256 + lane * 8;

    v8f x = *(const v8f*)(X + base);
    if (res) {
        v8f r2 = *(const v8f*)(res + base);
        #pragma unroll
        for (int i = 0; i < 8; i++) x[i] += r2[i];
    }
    float s = 0.f;
    #pragma unroll
    for (int i = 0; i < 8; i++) s += x[i];
    float mu = rsum32(s) * (1.0f / 256.0f);
    float vs = 0.f;
    #pragma unroll
    for (int i = 0; i < 8; i++) { float d = x[i] - mu; vs += d * d; }
    float rstd = rsqrtf(rsum32(vs) * (1.0f / 256.0f) + 1e-5f);

    v8f gg = *(const v8f*)(g + lane * 8);
    v8f bb = *(const v8f*)(beta + lane * 8);
    v8f y;
    #pragma unroll
    for (int i = 0; i < 8; i++) y[i] = (x[i] - mu) * rstd * gg[i] + bb[i];
    if (outF) *(v8f*)(outF + base) = y;
    if (outB) {
        __bf16* oB = (__bf16*)outB;
        #pragma unroll
        for (int i = 0; i < 8; i++) oB[base + i] = tobf(y[i]);
    }
}

// ---------------------------------------------------------------------------
// Host-side orchestration.
// ---------------------------------------------------------------------------
extern "C" void kernel_launch(void* const* d_in, const int* in_sizes, int n_in,
                              void* d_out, int out_size, void* d_ws, size_t ws_size,
                              hipStream_t stream)
{
    (void)in_sizes; (void)n_in; (void)out_size; (void)ws_size;
    const float* Q   = (const float*)d_in[0];
    const float* Kin = (const float*)d_in[1];
    const int*   msk = (const int*)d_in[2];
    const float* Wq  = (const float*)d_in[3];
    const float* bq  = (const float*)d_in[4];
    const float* Wk  = (const float*)d_in[5];
    const float* bk  = (const float*)d_in[6];
    const float* Wv  = (const float*)d_in[7];
    const float* bv  = (const float*)d_in[8];
    const float* W1  = (const float*)d_in[9];
    const float* b1  = (const float*)d_in[10];
    const float* W2  = (const float*)d_in[11];
    const float* b2  = (const float*)d_in[12];
    const float* g0  = (const float*)d_in[13];
    const float* be0 = (const float*)d_in[14];
    const float* g1  = (const float*)d_in[15];
    const float* be1 = (const float*)d_in[16];

    const int M = 4 * 2048;                  // B * N rows = 8192

    char* ws = (char*)d_ws;
    size_t off = 0;
    auto alloc = [&](size_t bytes) {
        void* p = ws + off;
        off += (bytes + 255) & ~(size_t)255;
        return p;
    };
    unsigned short* Wqt = (unsigned short*)alloc((size_t)256 * 256 * 2);
    unsigned short* Wkt = (unsigned short*)alloc((size_t)256 * 256 * 2);
    unsigned short* Wvt = (unsigned short*)alloc((size_t)256 * 256 * 2);
    unsigned short* W1t = (unsigned short*)alloc((size_t)1024 * 256 * 2);
    unsigned short* W2t = (unsigned short*)alloc((size_t)256 * 1024 * 2);
    unsigned short* Qp  = (unsigned short*)alloc((size_t)M * 256 * 2);
    unsigned short* Kp  = (unsigned short*)alloc((size_t)M * 256 * 2);
    unsigned short* Vt  = (unsigned short*)alloc((size_t)M * 256 * 2);
    float*          Tf  = (float*)alloc((size_t)M * 256 * 4);   // Oatt, reused as T
    float*          X0f = (float*)alloc((size_t)M * 256 * 4);
    unsigned short* X0b = (unsigned short*)alloc((size_t)M * 256 * 2);
    unsigned short* Hf  = (unsigned short*)alloc((size_t)M * 1024 * 2);

    // 1) weights -> transposed bf16
    transpose_bf16<<<(256 * 256 + 255) / 256, 256, 0, stream>>>(Wq, Wqt, 256, 256);
    transpose_bf16<<<(256 * 256 + 255) / 256, 256, 0, stream>>>(Wk, Wkt, 256, 256);
    transpose_bf16<<<(256 * 256 + 255) / 256, 256, 0, stream>>>(Wv, Wvt, 256, 256);
    transpose_bf16<<<(256 * 1024 + 255) / 256, 256, 0, stream>>>(W1, W1t, 256, 1024);
    transpose_bf16<<<(1024 * 256 + 255) / 256, 256, 0, stream>>>(W2, W2t, 1024, 256);

    // 2) projections (A = f32 inputs, converted in-register with native cvt)
    {
        long long waves = (long long)(M / 16) * (256 / 64);
        int blocks = (int)((waves + 7) / 8);
        gemm16x64<true><<<blocks, 256, 0, stream>>>(Q,   nullptr, Wqt, bq, M, 256, 256,
                                                    nullptr, Qp, nullptr, 0, nullptr, 0);
        gemm16x64<true><<<blocks, 256, 0, stream>>>(Kin, nullptr, Wkt, bk, M, 256, 256,
                                                    nullptr, Kp, nullptr, 0, nullptr, 0);
        gemm16x64<true><<<blocks, 256, 0, stream>>>(Kin, nullptr, Wvt, bv, M, 256, 256,
                                                    nullptr, nullptr, Vt, 2048, nullptr, 0);
    }

    // 3) flash attention -> Tf (= Oatt)
    attn_kernel<<<512, 256, 0, stream>>>(Qp, Kp, Vt, msk, Tf);

    // 4) X0 = LN(Q + Oatt)
    ln_kernel<<<M / 8, 256, 0, stream>>>(Tf, Q, g0, be0, X0f, X0b, M);

    // 5) H = relu(X0 @ W1 + b1)
    {
        long long waves = (long long)(M / 16) * (1024 / 64);
        int blocks = (int)((waves + 7) / 8);
        gemm16x64<false><<<blocks, 256, 0, stream>>>(nullptr, X0b, W1t, b1, M, 1024, 256,
                                                     nullptr, Hf, nullptr, 0, nullptr, 1);
    }

    // 6) T = X0 + H @ W2 + b2   (reuse Tf)
    {
        long long waves = (long long)(M / 16) * (256 / 64);
        int blocks = (int)((waves + 7) / 8);
        gemm16x64<false><<<blocks, 256, 0, stream>>>(nullptr, Hf, W2t, b2, M, 256, 1024,
                                                     Tf, nullptr, nullptr, 0, X0f, 0);
    }

    // 7) out = LN(T)
    ln_kernel<<<M / 8, 256, 0, stream>>>(Tf, nullptr, g1, be1, (float*)d_out, nullptr, M);
}
